// CertifiedTemporalAttention_29549374996666
// MI455X (gfx1250) — matrix operand order repaired and checked by reading, not compiled, not run on hardware
//
#include <hip/hip_runtime.h>
#include <hip/hip_bf16.h>

typedef __attribute__((ext_vector_type(16))) _Float16 v16h;
typedef __attribute__((ext_vector_type(8)))  float    v8f;
typedef __attribute__((ext_vector_type(4)))  unsigned su4;
typedef __attribute__((ext_vector_type(8)))  unsigned su8;

#define L_SEQ 2048
#define DIM   512
#define NH    8
#define HDIM  64
#define NB    2

__device__ inline v8f wmma_f16f32(v16h a, v16h b, v8f c) {
  // 8 args: (neg_a, A, neg_b, B, c_mod, C, reuse_a, reuse_b)
  return __builtin_amdgcn_wmma_f32_16x16x32_f16(false, a, false, b, (short)0, c, false, false);
}

// ---------------------------------------------------------------------------
// Kernel 1: LayerNorm over D=512, fp32 in -> f16 out. One block (128 thr) / row.
// ---------------------------------------------------------------------------
__global__ __launch_bounds__(128) void cta_ln_kernel(const float* __restrict__ x,
                                                     const float* __restrict__ gamma,
                                                     const float* __restrict__ beta,
                                                     _Float16* __restrict__ y) {
  __shared__ float red[128];
  const int row = blockIdx.x;
  const int t   = threadIdx.x;
  const float* xr = x + (size_t)row * DIM;

  float s = 0.f;
  for (int i = t; i < DIM; i += 128) s += xr[i];
  red[t] = s; __syncthreads();
  for (int off = 64; off > 0; off >>= 1) { if (t < off) red[t] += red[t + off]; __syncthreads(); }
  const float mu = red[0] * (1.0f / DIM);
  __syncthreads();

  float s2 = 0.f;
  for (int i = t; i < DIM; i += 128) { float d = xr[i] - mu; s2 += d * d; }
  red[t] = s2; __syncthreads();
  for (int off = 64; off > 0; off >>= 1) { if (t < off) red[t] += red[t + off]; __syncthreads(); }
  const float rstd = rsqrtf(red[0] * (1.0f / DIM) + 1e-5f);

  for (int i = t; i < DIM; i += 128)
    y[(size_t)row * DIM + i] = (_Float16)((xr[i] - mu) * rstd * gamma[i] + beta[i]);
}

// ---------------------------------------------------------------------------
// Kernel 2: fp32 -> f16 convert (for Wq/Wk/Wv)
// ---------------------------------------------------------------------------
__global__ __launch_bounds__(256) void cta_cvt_kernel(const float* __restrict__ in,
                                                      _Float16* __restrict__ out, int n) {
  int i = blockIdx.x * 256 + threadIdx.x;
  if (i < n) out[i] = (_Float16)in[i];
}

// ---------------------------------------------------------------------------
// Kernel 3: QKV projection GEMM via WMMA. One wave computes a 16x64 tile.
//   grid = (rows/16=256, cols/64=8, 3)   z: 0=Q, 1=K, 2=V(transposed store)
//   Q,K layout [B,H,L,64]; Vt layout [B,H,64,L]
// ---------------------------------------------------------------------------
__global__ __launch_bounds__(32) void cta_qkv_gemm(const _Float16* __restrict__ X,
                                                   const _Float16* __restrict__ Wq,
                                                   const _Float16* __restrict__ Wk,
                                                   const _Float16* __restrict__ Wv,
                                                   _Float16* __restrict__ Qo,
                                                   _Float16* __restrict__ Ko,
                                                   _Float16* __restrict__ VTo) {
  const int tM = blockIdx.x, tN = blockIdx.y, z = blockIdx.z;
  const _Float16* W = (z == 0) ? Wq : (z == 1) ? Wk : Wv;
  const int lane = threadIdx.x & 31;
  const int n = lane & 15, hi = lane >> 4;
  const int r0 = tM * 16;

  v8f acc[4];
  v8f zero = {};
#pragma unroll
  for (int t = 0; t < 4; ++t) acc[t] = zero;

  for (int k0 = 0; k0 < DIM; k0 += 32) {
    // A fragment: lane = M row, halves = K {16*hi .. +15} contiguous
    v16h a;
    const _Float16* p = X + (size_t)(r0 + n) * DIM + k0 + 16 * hi;
#pragma unroll
    for (int i = 0; i < 8; ++i) { a[i] = p[i]; a[i + 8] = p[i + 8]; }
    __builtin_prefetch(p + 32, 0, 3);   // global_prefetch_b8 next k-step
#pragma unroll
    for (int t = 0; t < 4; ++t) {
      // B fragment: lane = N col, halves = K {16*hi .. +15} contiguous
      const v16h b = *(const v16h*)(W + (size_t)(tN * 64 + t * 16 + n) * DIM + k0 + 16 * hi);
      acc[t] = wmma_f16f32(a, b, acc[t]);
    }
  }

#pragma unroll
  for (int t = 0; t < 4; ++t) {
#pragma unroll
    for (int g = 0; g < 8; ++g) {
      const int lg = r0 + g + 8 * hi;        // global row 0..4095
      const int b  = lg >> 11, ll = lg & (L_SEQ - 1);
      const int o  = tN * 64 + t * 16 + n;
      const int h  = o >> 6, hd = o & 63;
      const _Float16 v = (_Float16)acc[t][g];
      if (z == 0)      Qo[(((size_t)(b * NH + h)) * L_SEQ + ll) * HDIM + hd] = v;
      else if (z == 1) Ko[(((size_t)(b * NH + h)) * L_SEQ + ll) * HDIM + hd] = v;
      else             VTo[(((size_t)(b * NH + h)) * HDIM + hd) * L_SEQ + ll] = v;
    }
  }
}

// ---------------------------------------------------------------------------
// Kernel 4: banded flash attention. One wave per (b, h, 16-query tile).
//   grid = B*H*(L/16) = 2048 blocks of 32 threads.
//   K chunks staged via Tensor Data Mover (tensor_load_to_lds, TENSORcnt),
//   V chunks staged via global_load_async_to_lds_b128 (ASYNCcnt),
//   double-buffered so one chunk's DMA is in flight behind the WMMAs.
// ---------------------------------------------------------------------------
__global__ __launch_bounds__(32) void cta_attn_kernel(const _Float16* __restrict__ Q,
                                                      const _Float16* __restrict__ K,
                                                      const _Float16* __restrict__ Vt,
                                                      const float* __restrict__ ts,
                                                      const int* __restrict__ lens,
                                                      const float* __restrict__ tw,
                                                      _Float16* __restrict__ att) {
  __shared__ _Float16 kS[2][32 * HDIM];   // K chunk: [32 keys][64 d], double buffered
  __shared__ _Float16 vS[2][HDIM * 32];   // V chunk: [64 d][32 keys], double buffered
  __shared__ _Float16 pS[16 * 32];        // P tile staging (C-layout -> A-layout)

  const int id = blockIdx.x;
  const int qt = id & 127, h = (id >> 7) & 7, b = id >> 10;
  const int q0 = qt * 16;
  const int lane = threadIdx.x & 31;
  const int n = lane & 15, hi = lane >> 4;

  const size_t bh = (size_t)(b * NH + h);
  const _Float16* Qb = Q  + bh * L_SEQ * HDIM;
  const _Float16* Kb = K  + bh * L_SEQ * HDIM;
  const _Float16* Vb = Vt + bh * HDIM * L_SEQ;
  const float* tsb = ts + (size_t)b * L_SEQ;
  const float aw = fabsf(tw[0]);
  const int len = lens[b];

  const unsigned lbaseK = (unsigned)(size_t)(&kS[0][0]);
  const unsigned lbaseV = (unsigned)(size_t)(&vS[0][0]);

  // ---- chunk stager: TDM for K tile, async global->LDS for V tile ----
  auto stage = [&](int kcS, int buf) {
    // K tile [32 rows x 64 halves] via Tensor Data Mover.
    // D# group0: count=1 | lds_addr | global_addr(57b) | type=2
    const unsigned long long ga =
        (unsigned long long)(size_t)(Kb + (size_t)kcS * HDIM);
    su4 g0;
    g0[0] = 1u;                                      // count = 1 descriptor
    g0[1] = lbaseK + (unsigned)(buf * 32 * HDIM * 2);
    g0[2] = (unsigned)(ga & 0xffffffffu);
    g0[3] = (unsigned)((ga >> 32) & 0x01ffffffu) | (2u << 30);
    // D# group1: wg_mask=0, data_size=2B, dims: tensor 64x32, tile 64x32, stride 64
    su8 g1;
    g1[0] = 0x00010000u;        // data_size = 1 (2 bytes)
    g1[1] = (64u << 16);        // tensor_dim0[15:0]  = 64 (bits 63:48)
    g1[2] = (32u << 16);        // tensor_dim0 hi = 0 | tensor_dim1[15:0] = 32
    g1[3] = (64u << 16);        // tensor_dim1 hi = 0 | tile_dim0 = 64
    g1[4] = 32u;                // tile_dim1 = 32, tile_dim2 = 0
    g1[5] = 64u;                // tensor_dim0_stride = 64 (low 32)
    g1[6] = 0u;
    g1[7] = 0u;
    asm volatile("tensor_load_to_lds %0, %1" :: "s"(g0), "s"(g1) : "memory");

    // V tile: 64 rows x 32 halves (64B) -> 256 x 16B segments, 8 per lane
    const unsigned vb = lbaseV + (unsigned)(buf * HDIM * 32 * 2);
#pragma unroll
    for (int it = 0; it < 8; ++it) {
      const int seg  = it * 32 + lane;
      const int d    = seg >> 2;
      const int part = seg & 3;
      const unsigned dst = vb + (unsigned)((d * 32 + part * 8) * 2);
      const _Float16* src = Vb + (size_t)d * L_SEQ + kcS + part * 8;
      asm volatile("global_load_async_to_lds_b128 %0, %1, off"
                   :: "v"(dst), "v"(src) : "memory");
    }
  };

  // Q A-fragments (16 x 64 in two k=32 chunks)
  v16h aQ[2];
#pragma unroll
  for (int kk = 0; kk < 2; ++kk) {
    const _Float16* p = Qb + (size_t)(q0 + n) * HDIM + kk * 32 + 16 * hi;
#pragma unroll
    for (int i = 0; i < 8; ++i) { aQ[kk][i] = p[i]; aQ[kk][i + 8] = p[i + 8]; }
  }
  float tq[8];
#pragma unroll
  for (int g = 0; g < 8; ++g) tq[g] = tsb[q0 + g + 8 * hi];

  float m[8], l[8];
  v8f O[4];
  v8f zero = {};
#pragma unroll
  for (int g = 0; g < 8; ++g) { m[g] = -80.0f; l[g] = 0.0f; }
#pragma unroll
  for (int t = 0; t < 4; ++t) O[t] = zero;

  const int cLo = (q0 - 128 > 0 ? (q0 - 128) : 0) >> 5;
  const int cHiK = (q0 + 143 < L_SEQ - 1 ? q0 + 143 : L_SEQ - 1);
  const int cHi = cHiK >> 5;

  stage(cLo * 32, 0);                       // prologue: first chunk in flight

  for (int c = cLo; c <= cHi; ++c) {
    const int kc = c * 32;
    const int buf = (c - cLo) & 1;
    if (c < cHi) stage((c + 1) * 32, buf ^ 1);   // keep next chunk in flight

    if (c < cHi) {
      __builtin_amdgcn_s_wait_tensorcnt(1);      // this chunk's TDM done
      asm volatile("s_wait_asynccnt 0x8" ::: "memory");
    } else {
      __builtin_amdgcn_s_wait_tensorcnt(0);
      asm volatile("s_wait_asynccnt 0x0" ::: "memory");
    }
    __syncthreads();

    const _Float16* Ksl = kS[buf];
    const _Float16* Vsl = vS[buf];

    v8f s[2];
#pragma unroll
    for (int sub = 0; sub < 2; ++sub) {
      v8f sacc = zero;
#pragma unroll
      for (int kk = 0; kk < 2; ++kk) {
        // K^T B-fragment from LDS: lane = key (N), halves = d {16*hi..+15}
        const v16h bK = *(const v16h*)(Ksl + (sub * 16 + n) * HDIM + kk * 32 + 16 * hi);
        sacc = wmma_f16f32(aQ[kk], bK, sacc);
      }
      const int kcol = kc + sub * 16 + n;
      const float tk = tsb[kcol];
      const bool pvalid = (kcol < len);
#pragma unroll
      for (int g = 0; g < 8; ++g) {
        const int qrow = q0 + g + 8 * hi;
        const int dq = qrow - kcol;
        const bool ok = pvalid && (dq <= 128) && (dq >= -128);
        const float bias = __logf(__expf(-aw * fabsf(tq[g] - tk)) + 1e-8f);
        s[sub][g] = ok ? (sacc[g] * 0.125f + bias) : -1e30f;
      }
    }

    // Online softmax update (row reductions across N = lanes within 16-group)
    float alpha[8];
#pragma unroll
    for (int g = 0; g < 8; ++g) {
      float r = fmaxf(s[0][g], s[1][g]);
      r = fmaxf(r, __shfl_xor(r, 1, 32));
      r = fmaxf(r, __shfl_xor(r, 2, 32));
      r = fmaxf(r, __shfl_xor(r, 4, 32));
      r = fmaxf(r, __shfl_xor(r, 8, 32));
      const float mn = fmaxf(m[g], r);
      alpha[g] = __expf(m[g] - mn);
      m[g] = mn;
      const float p0 = __expf(s[0][g] - mn);
      const float p1 = __expf(s[1][g] - mn);
      s[0][g] = p0; s[1][g] = p1;
      float rs = p0 + p1;
      rs += __shfl_xor(rs, 1, 32);
      rs += __shfl_xor(rs, 2, 32);
      rs += __shfl_xor(rs, 4, 32);
      rs += __shfl_xor(rs, 8, 32);
      l[g] = l[g] * alpha[g] + rs;
    }
#pragma unroll
    for (int t = 0; t < 4; ++t)
#pragma unroll
      for (int g = 0; g < 8; ++g) O[t][g] *= alpha[g];

    // Stage P (f16) to LDS in row-major [16][32], reload as A-fragment
#pragma unroll
    for (int g = 0; g < 8; ++g) {
      const int row = g + 8 * hi;
      pS[row * 32 + n]      = (_Float16)s[0][g];
      pS[row * 32 + 16 + n] = (_Float16)s[1][g];
    }
    __syncthreads();
    v16h aP;
    {
      const _Float16* p = &pS[n * 32 + 16 * hi];
#pragma unroll
      for (int i = 0; i < 8; ++i) { aP[i] = p[i]; aP[i + 8] = p[i + 8]; }
    }
    __syncthreads();

#pragma unroll
    for (int t = 0; t < 4; ++t) {
      // V^T B-fragment from LDS: lane = d (N), halves = key {16*hi..+15}
      const v16h bV = *(const v16h*)(Vsl + (t * 16 + n) * 32 + 16 * hi);
      O[t] = wmma_f16f32(aP, bV, O[t]);
    }
  }

  // Epilogue: divide by row sum, store attended [B, L, H*HD] as f16
#pragma unroll
  for (int g = 0; g < 8; ++g) {
    const float inv = 1.0f / l[g];
    const int row = q0 + g + 8 * hi;
#pragma unroll
    for (int t = 0; t < 4; ++t)
      att[((size_t)(b * L_SEQ + row)) * DIM + h * HDIM + t * 16 + n] = (_Float16)(O[t][g] * inv);
  }
}

// ---------------------------------------------------------------------------
// Kernel 5: position-weighted sums.  wa = sum_q pw_n * attended,
//           wr = sum_q pw_n * residual,  c = sum_q pw_n.   grid = (B), 256 thr
// ---------------------------------------------------------------------------
__global__ __launch_bounds__(256) void cta_agg_kernel(const _Float16* __restrict__ att,
                                                      const float* __restrict__ seq,
                                                      const int* __restrict__ lens,
                                                      float* __restrict__ wa,
                                                      float* __restrict__ wr,
                                                      float* __restrict__ cfac) {
  const int b = blockIdx.x, t = threadIdx.x;
  const int len = lens[b];
  float a0 = 0.f, a1 = 0.f, r0 = 0.f, r1 = 0.f, S = 0.f;
  for (int q = 0; q < L_SEQ; ++q) {
    const float pw = (q < len) ? __expf(0.1f * (float)(q - (L_SEQ - 1))) : 0.f;
    S += pw;
    const _Float16* ar = att + ((size_t)b * L_SEQ + q) * DIM;
    const float*    rr = seq + ((size_t)b * L_SEQ + q) * DIM;
    a0 += pw * (float)ar[t];       a1 += pw * (float)ar[t + 256];
    r0 += pw * rr[t];              r1 += pw * rr[t + 256];
  }
  const float inv = 1.f / (S + 1e-8f);
  wa[b * DIM + t] = a0 * inv;  wa[b * DIM + t + 256] = a1 * inv;
  wr[b * DIM + t] = r0 * inv;  wr[b * DIM + t + 256] = r1 * inv;
  if (t == 0) cfac[b] = S * inv;
}

// ---------------------------------------------------------------------------
// Kernel 6: final GEMV  out[b,o] = wa[b]·Wo[o,:] + c[b]*bo[o] + wr[b,o]
//   grid = (B, 2), 256 thr
// ---------------------------------------------------------------------------
__global__ __launch_bounds__(256) void cta_outproj_kernel(const float* __restrict__ wa,
                                                          const float* __restrict__ wr,
                                                          const float* __restrict__ cfac,
                                                          const float* __restrict__ Wo,
                                                          const float* __restrict__ bo,
                                                          float* __restrict__ out) {
  const int b = blockIdx.x;
  const int o = blockIdx.y * 256 + threadIdx.x;
  const float* w = Wo + (size_t)o * DIM;
  const float* a = wa + (size_t)b * DIM;
  float s = 0.f;
  for (int d = 0; d < DIM; ++d) s += a[d] * w[d];
  out[b * DIM + o] = s + cfac[b] * bo[o] + wr[b * DIM + o];
}

// ---------------------------------------------------------------------------
extern "C" void kernel_launch(void* const* d_in, const int* in_sizes, int n_in,
                              void* d_out, int out_size, void* d_ws, size_t ws_size,
                              hipStream_t stream) {
  (void)in_sizes; (void)n_in; (void)out_size; (void)ws_size;
  const float* seq   = (const float*)d_in[0];
  const int*   lens  = (const int*)  d_in[1];
  const float* ts    = (const float*)d_in[2];
  const float* gamma = (const float*)d_in[3];
  const float* beta  = (const float*)d_in[4];
  const float* Wq    = (const float*)d_in[5];
  const float* Wk    = (const float*)d_in[6];
  const float* Wv    = (const float*)d_in[7];
  const float* Wo    = (const float*)d_in[8];
  const float* bo    = (const float*)d_in[9];
  const float* tw    = (const float*)d_in[10];
  float* out = (float*)d_out;

  // workspace carve-up (256B aligned regions)
  char* ws = (char*)d_ws;
  size_t off = 0;
  auto take = [&](size_t bytes) { char* p = ws + off; off += (bytes + 255) & ~(size_t)255; return p; };
  _Float16* x16  = (_Float16*)take((size_t)NB * L_SEQ * DIM * sizeof(_Float16));
  _Float16* Wq16 = (_Float16*)take((size_t)DIM * DIM * sizeof(_Float16));
  _Float16* Wk16 = (_Float16*)take((size_t)DIM * DIM * sizeof(_Float16));
  _Float16* Wv16 = (_Float16*)take((size_t)DIM * DIM * sizeof(_Float16));
  _Float16* Q16  = (_Float16*)take((size_t)NB * L_SEQ * DIM * sizeof(_Float16));
  _Float16* K16  = (_Float16*)take((size_t)NB * L_SEQ * DIM * sizeof(_Float16));
  _Float16* Vt16 = (_Float16*)take((size_t)NB * L_SEQ * DIM * sizeof(_Float16));
  _Float16* att  = (_Float16*)take((size_t)NB * L_SEQ * DIM * sizeof(_Float16));
  float* wa   = (float*)take((size_t)NB * DIM * sizeof(float));
  float* wr   = (float*)take((size_t)NB * DIM * sizeof(float));
  float* cfac = (float*)take((size_t)NB * sizeof(float));

  // 1. LayerNorm -> f16
  cta_ln_kernel<<<NB * L_SEQ, 128, 0, stream>>>(seq, gamma, beta, x16);

  // 2. weight conversion
  const int wn = DIM * DIM;
  cta_cvt_kernel<<<(wn + 255) / 256, 256, 0, stream>>>(Wq, Wq16, wn);
  cta_cvt_kernel<<<(wn + 255) / 256, 256, 0, stream>>>(Wk, Wk16, wn);
  cta_cvt_kernel<<<(wn + 255) / 256, 256, 0, stream>>>(Wv, Wv16, wn);

  // 3. QKV GEMMs (WMMA)
  cta_qkv_gemm<<<dim3(NB * L_SEQ / 16, DIM / 64, 3), 32, 0, stream>>>(
      x16, Wq16, Wk16, Wv16, Q16, K16, Vt16);

  // 4. banded flash attention (WMMA + TDM + async LDS copies)
  cta_attn_kernel<<<NB * NH * (L_SEQ / 16), 32, 0, stream>>>(
      Q16, K16, Vt16, ts, lens, tw, att);

  // 5. position-weighted aggregation
  cta_agg_kernel<<<NB, 256, 0, stream>>>(att, seq, lens, wa, wr, cfac);

  // 6. final projection + bias + residual aggregate
  cta_outproj_kernel<<<dim3(NB, DIM / 256), 256, 0, stream>>>(wa, wr, cfac, Wo, bo, out);
}
